// GConvGRUModel_2448131359039
// MI455X (gfx1250) — compile-verified, matrix operand order checked by reading, not run on hardware
//
#include <hip/hip_runtime.h>
#include <math.h>

// ---------------------------------------------------------------------------
// GConvGRU (h0 = 0) on gfx1250:
//   * sparse Cheb propagation: edge-parallel b128 gathers + native fp32
//     atomics resolved in L2 (192MB L2 holds the working set)
//   * dense gate GEMMs: v_wmma_f32_16x16x32_f16 on zero-padded activations and
//     pre-swizzled f16 weight fragments -> branch-free b128 inner loop
//   * gate epilogue on hardware v_tanh_f32 / v_exp_f32
// ---------------------------------------------------------------------------

typedef __attribute__((ext_vector_type(16))) _Float16 v16h;
typedef __attribute__((ext_vector_type(8)))  float    v8f;

__device__ __forceinline__ void atomAddF(float* p, float v) {
  __hip_atomic_fetch_add(p, v, __ATOMIC_RELAXED, __HIP_MEMORY_SCOPE_AGENT);
}

__device__ __forceinline__ float fast_tanh(float x) {
#if __has_builtin(__builtin_amdgcn_tanhf)
  return __builtin_amdgcn_tanhf(x);          // v_tanh_f32 (CDNA5 TRANS op)
#else
  float e = __expf(2.0f * x);                // v_exp_f32 path
  return 1.0f - 2.0f / (e + 1.0f);
#endif
}

// y[i] = (negsrc ? -negsrc[i] : 0)
__global__ void k_init(float* __restrict__ y, const float* __restrict__ negsrc, int n) {
  int i = blockIdx.x * blockDim.x + threadIdx.x;
  if (i < n) y[i] = negsrc ? -negsrc[i] : 0.0f;
}

// y[r*finp + c] = (c < fin) ? x[r*fin + c] : 0
__global__ void k_copy_pad(const float* __restrict__ x, float* __restrict__ y,
                           int n, int fin, int finp) {
  int i = blockIdx.x * blockDim.x + threadIdx.x;
  if (i >= n) return;
  int r = i / finp, c = i - r * finp;
  y[i] = (c < fin) ? x[(size_t)r * fin + c] : 0.0f;
}

__global__ void k_deg(const int* __restrict__ row, float* __restrict__ deg, int E) {
  int e = blockIdx.x * blockDim.x + threadIdx.x;
  if (e < E) atomAddF(&deg[row[e]], 1.0f);
}

__global__ void k_dinv(float* __restrict__ deg, int n) {
  int i = blockIdx.x * blockDim.x + threadIdx.x;
  if (i < n) {
    float d = deg[i];
    deg[i] = (d > 0.0f) ? rsqrtf(d) : 0.0f;
  }
}

__global__ void k_edge_w(const int* __restrict__ row, const int* __restrict__ col,
                         const float* __restrict__ dinv, float* __restrict__ w, int E) {
  int e = blockIdx.x * blockDim.x + threadIdx.x;
  if (e < E) w[e] = -dinv[row[e]] * dinv[col[e]];
}

// y[dst*stride + f] += coef * w[e] * x[src*stride + f], f < fin (scalar loads)
__global__ void k_prop(const float* __restrict__ x, float* __restrict__ y,
                       const float* __restrict__ w,
                       const int* __restrict__ src, const int* __restrict__ dst,
                       int E, int fin, int stride, float coef) {
  int e = blockIdx.x * blockDim.x + threadIdx.x;
  if (e >= E) return;
  float wc = coef * w[e];
  const float* xs = x + (size_t)src[e] * stride;
  float* yd = y + (size_t)dst[e] * stride;
  for (int f = 0; f < fin; ++f) atomAddF(&yd[f], wc * xs[f]);
}

// Same, fin % 4 == 0: b128 gathers (layers 1-3, the bulk of scatter traffic)
template <int FIN>
__global__ void k_prop4(const float* __restrict__ x, float* __restrict__ y,
                        const float* __restrict__ w,
                        const int* __restrict__ src, const int* __restrict__ dst,
                        int E, int stride, float coef) {
  int e = blockIdx.x * blockDim.x + threadIdx.x;
  if (e >= E) return;
  float wc = coef * w[e];
  const float4* xs = (const float4*)(x + (size_t)src[e] * stride);
  float* yd = y + (size_t)dst[e] * stride;
#pragma unroll
  for (int q = 0; q < FIN / 4; ++q) {
    float4 v = xs[q];
    atomAddF(&yd[q * 4 + 0], wc * v.x);
    atomAddF(&yd[q * 4 + 1], wc * v.y);
    atomAddF(&yd[q * 4 + 2], wc * v.z);
    atomAddF(&yd[q * 4 + 3], wc * v.w);
  }
}

// Pack W (K, fin, fout) fp32 -> f16 fragments in the exact per-lane B layout:
// out[(((k*chunks + c)*ctiles + ct)*32 + lane)*16 + h]
__global__ void k_pack_w(const float* __restrict__ W, _Float16* __restrict__ out,
                         int fin, int fout, int K, int chunks, int ctiles) {
  int idx = blockIdx.x * blockDim.x + threadIdx.x;
  int total = K * chunks * ctiles * 32 * 16;
  if (idx >= total) return;
  int h = idx & 15;
  int lane = (idx >> 4) & 31;
  int t = idx >> 9;
  int ct = t % ctiles; t /= ctiles;
  int c  = t % chunks;
  int k  = t / chunks;
  int khalf = lane >> 4;
  int kloc = (h & 7) + khalf * 8 + ((h >= 8) ? 16 : 0);
  int kg = c * 32 + kloc;
  int col = ct * 16 + (lane & 15);
  float v = (kg < fin && col < fout) ? W[((size_t)k * fin + kg) * fout + col] : 0.0f;
  out[idx] = (_Float16)v;
}

struct TxPtrs { const float* p[5]; };

// One wave per 16x16 output tile over the padded SOUT columns.
// Branch-free inner loop: 4x b128 A loads + packed cvt, 2x 32B packed-B loads,
// 2x v_wmma_f32_16x16x32_f16 (z-path, h-path sharing the A fragment).
template <int FIN, int FINP, int FOUT, int SOUT, int K>
__global__ void k_layer_wmma(TxPtrs tx,
                             const _Float16* __restrict__ PBz,
                             const _Float16* __restrict__ PBh,
                             const float* __restrict__ bxz, const float* __restrict__ bhz,
                             const float* __restrict__ bxh, const float* __restrict__ bhh,
                             float* __restrict__ hout, int N) {
  constexpr int CHUNKS = FINP / 32;
  constexpr int CTILES = (SOUT + 15) / 16;
  const int wavesPerBlock = blockDim.x >> 5;
  const int waveId = blockIdx.x * wavesPerBlock + (threadIdx.x >> 5);
  const int rowTiles = N >> 4;                 // N % 16 == 0
  if (waveId >= rowTiles * CTILES) return;
  const int rt = waveId / CTILES;
  const int ct = waveId % CTILES;
  const int lane  = threadIdx.x & 31;
  const int khalf = lane >> 4;
  const int row   = rt * 16 + (lane & 15);
  const int col   = ct * 16 + (lane & 15);

  v8f accz = {};
  v8f acch = {};

#pragma unroll
  for (int k = 0; k < K; ++k) {
    const float* __restrict__ Ak = tx.p[k] + (size_t)row * FINP + khalf * 8;
#pragma unroll
    for (int c = 0; c < CHUNKS; ++c) {
      const float4* ap0 = (const float4*)(Ak + c * 32);
      const float4* ap1 = (const float4*)(Ak + c * 32 + 16);
      float4 a0 = ap0[0], a1 = ap0[1];         // K = c*32 + khalf*8 + 0..7
      float4 a2 = ap1[0], a3 = ap1[1];         // K = c*32 + 16 + khalf*8 + 0..7
      v16h a;
      a[0]  = (_Float16)a0.x; a[1]  = (_Float16)a0.y;
      a[2]  = (_Float16)a0.z; a[3]  = (_Float16)a0.w;
      a[4]  = (_Float16)a1.x; a[5]  = (_Float16)a1.y;
      a[6]  = (_Float16)a1.z; a[7]  = (_Float16)a1.w;
      a[8]  = (_Float16)a2.x; a[9]  = (_Float16)a2.y;
      a[10] = (_Float16)a2.z; a[11] = (_Float16)a2.w;
      a[12] = (_Float16)a3.x; a[13] = (_Float16)a3.y;
      a[14] = (_Float16)a3.z; a[15] = (_Float16)a3.w;

      const size_t bidx = ((((size_t)k * CHUNKS + c) * CTILES + ct) * 32 + lane) * 16;
      v16h bz = *(const v16h*)(PBz + bidx);
      v16h bh = *(const v16h*)(PBh + bidx);

      accz = __builtin_amdgcn_wmma_f32_16x16x32_f16(false, a, false, bz,
                                                    (short)0, accz, false, false);
      acch = __builtin_amdgcn_wmma_f32_16x16x32_f16(false, a, false, bh,
                                                    (short)0, acch, false, false);
    }
  }

  // Gate epilogue: z = sigmoid(gz + b), h~ = tanh(gh + b), out = relu((1-z)h~)
  const bool live = (col < FOUT);
  const float bzc = live ? (bxz[col] + bhz[col]) : 0.0f;
  const float bhc = live ? (bxh[col] + bhh[col]) : 0.0f;
#pragma unroll
  for (int i = 0; i < 8; ++i) {
    int r = rt * 16 + khalf * 8 + i;           // C/D: vgpr i -> M = i (+8 hi lanes)
    float z  = 1.0f / (1.0f + __expf(-(accz[i] + bzc)));
    float ht = fast_tanh(acch[i] + bhc);
    float o  = (1.0f - z) * ht;
    o = (o > 0.0f) ? o : 0.0f;
    if (col < SOUT) hout[(size_t)r * SOUT + col] = live ? o : 0.0f;
  }
}

// logits = h @ linW + linb ; softmax over 2 classes (F = 152, 4-wide)
__global__ void k_head(const float* __restrict__ h, const float* __restrict__ W,
                       const float* __restrict__ b, float* __restrict__ out, int N) {
  int i = blockIdx.x * blockDim.x + threadIdx.x;
  if (i >= N) return;
  const float4*  hp = (const float4*)(h + (size_t)i * 152);
  const float2*  wp = (const float2*)W;
  float l0 = b[0], l1 = b[1];
#pragma unroll 4
  for (int q = 0; q < 38; ++q) {
    float4 v = hp[q];
    float2 w0 = wp[q * 4 + 0], w1 = wp[q * 4 + 1], w2 = wp[q * 4 + 2], w3 = wp[q * 4 + 3];
    l0 += v.x * w0.x + v.y * w1.x + v.z * w2.x + v.w * w3.x;
    l1 += v.x * w0.y + v.y * w1.y + v.z * w2.y + v.w * w3.y;
  }
  float mx = fmaxf(l0, l1);
  float e0 = __expf(l0 - mx), e1 = __expf(l1 - mx);
  float s = e0 + e1;
  out[(size_t)i * 2 + 0] = e0 / s;
  out[(size_t)i * 2 + 1] = e1 / s;
}

// ---------------------------------------------------------------------------

extern "C" void kernel_launch(void* const* d_in, const int* in_sizes, int n_in,
                              void* d_out, int out_size, void* d_ws, size_t ws_size,
                              hipStream_t stream) {
  const float* x    = (const float*)d_in[0];
  const int*   ei   = (const int*)d_in[1];
  const float* linW = (const float*)d_in[2];
  const float* linb = (const float*)d_in[3];
  const int N = in_sizes[0] / 10;
  const int E = in_sizes[1] / 2;
  const int* src = ei;
  const int* dst = ei + E;

  // ---- workspace carve (256B aligned) ----
  char* ws = (char*)d_ws;
  auto carve = [&](size_t bytes) -> char* {
    char* p = ws;
    ws += (bytes + 255) & ~(size_t)255;
    return p;
  };
  float*     wE     = (float*)carve((size_t)E * 4);
  float*     deg    = (float*)carve((size_t)N * 4);
  float*     xpad   = (float*)carve((size_t)N * 32 * 4);        // layer0 Tx0 (stride 32)
  float*     txpool = (float*)carve((size_t)N * 64 * 4 * 4);    // up to 4 Tx bufs, stride<=64
  float*     hA     = (float*)carve((size_t)N * 160 * 4);
  float*     hB     = (float*)carve((size_t)N * 160 * 4);
  _Float16*  pwZ    = (_Float16*)carve(256 * 1024);             // packed z-weights
  _Float16*  pwH    = (_Float16*)carve(256 * 1024);             // packed h-weights

  const int TB = 256;
  const int gE = (E + TB - 1) / TB;
  const int gN = (N + TB - 1) / TB;

  // ---- symmetric normalization: w = -dinv[src] * dinv[dst] ----
  k_init<<<gN, TB, 0, stream>>>(deg, nullptr, N);
  k_deg<<<gE, TB, 0, stream>>>(src, deg, E);
  k_dinv<<<gN, TB, 0, stream>>>(deg, N);
  k_edge_w<<<gE, TB, 0, stream>>>(src, dst, deg, wE, E);

  // ---- layer configs: FIN, FINP (in-stride), FOUT, SOUT (out-stride), K ----
  static const int LFIN[4]  = {10, 16, 32, 64};
  static const int LFINP[4] = {32, 32, 64, 64};
  static const int LFOUT[4] = {16, 32, 64, 152};
  static const int LSOUT[4] = {32, 64, 64, 152};
  static const int LK[4]    = {5, 4, 3, 2};

  // layer0 input: pad x (N x 10) into stride-32 buffer
  {
    int n = N * 32;
    k_copy_pad<<<(n + TB - 1) / TB, TB, 0, stream>>>(x, xpad, n, 10, 32);
  }

  const float* hprev = xpad;
  float* hbufs[2] = {hA, hB};

  for (int l = 0; l < 4; ++l) {
    const int fin = LFIN[l], finp = LFINP[l], fout = LFOUT[l], sout = LSOUT[l], K = LK[l];
    const int chunks = finp / 32;
    const int ctiles = (sout + 15) / 16;

    TxPtrs tx;
    tx.p[0] = hprev;
    for (int k = 1; k < 5; ++k) tx.p[k] = nullptr;

    // Cheb recurrence on padded buffers (pad columns stay zero).
    for (int k = 1; k < K; ++k) {
      float* buf = txpool + (size_t)(k - 1) * N * finp;
      tx.p[k] = buf;
      const float* neg = (k >= 2) ? tx.p[k - 2] : nullptr;
      int n = N * finp;
      k_init<<<(n + TB - 1) / TB, TB, 0, stream>>>(buf, neg, n);
      float coef = (k == 1) ? 1.0f : 2.0f;
      switch (fin) {
        case 16: k_prop4<16><<<gE, TB, 0, stream>>>(tx.p[k-1], buf, wE, src, dst, E, finp, coef); break;
        case 32: k_prop4<32><<<gE, TB, 0, stream>>>(tx.p[k-1], buf, wE, src, dst, E, finp, coef); break;
        case 64: k_prop4<64><<<gE, TB, 0, stream>>>(tx.p[k-1], buf, wE, src, dst, E, finp, coef); break;
        default: k_prop<<<gE, TB, 0, stream>>>(tx.p[k-1], buf, wE, src, dst, E, fin, finp, coef); break;
      }
    }

    // weights: base 4 + 12*l
    // [x_z_W, x_z_b, h_z_W, h_z_b, x_r_W, x_r_b, h_r_W, h_r_b, x_h_W, x_h_b, h_h_W, h_h_b]
    const int base = 4 + 12 * l;
    const float* Wz  = (const float*)d_in[base + 0];
    const float* bxz = (const float*)d_in[base + 1];
    const float* bhz = (const float*)d_in[base + 3];
    const float* Wh  = (const float*)d_in[base + 8];
    const float* bxh = (const float*)d_in[base + 9];
    const float* bhh = (const float*)d_in[base + 11];

    // pack weights into per-lane B fragments (f16)
    {
      int total = K * chunks * ctiles * 32 * 16;
      int g = (total + TB - 1) / TB;
      k_pack_w<<<g, TB, 0, stream>>>(Wz, pwZ, fin, fout, K, chunks, ctiles);
      k_pack_w<<<g, TB, 0, stream>>>(Wh, pwH, fin, fout, K, chunks, ctiles);
    }

    float* hout = hbufs[l & 1];
    const int tiles = (N / 16) * ctiles;
    const int wpb = TB / 32;
    const int gT = (tiles + wpb - 1) / wpb;
    switch (l) {
      case 0: k_layer_wmma<10, 32, 16, 32, 5><<<gT, TB, 0, stream>>>(
                  tx, pwZ, pwH, bxz, bhz, bxh, bhh, hout, N); break;
      case 1: k_layer_wmma<16, 32, 32, 64, 4><<<gT, TB, 0, stream>>>(
                  tx, pwZ, pwH, bxz, bhz, bxh, bhh, hout, N); break;
      case 2: k_layer_wmma<32, 64, 64, 64, 3><<<gT, TB, 0, stream>>>(
                  tx, pwZ, pwH, bxz, bhz, bxh, bhh, hout, N); break;
      case 3: k_layer_wmma<64, 64, 152, 152, 2><<<gT, TB, 0, stream>>>(
                  tx, pwZ, pwH, bxz, bhz, bxh, bhh, hout, N); break;
    }
    hprev = hout;
  }

  // ---- linear head + softmax ----
  k_head<<<gN, TB, 0, stream>>>(hprev, linW, linb, (float*)d_out, N);
}
